// SelfAttention_49228915146768
// MI455X (gfx1250) — matrix operand order repaired and checked by reading, not compiled, not run on hardware
//
#include <hip/hip_runtime.h>

typedef _Float16 v16h __attribute__((ext_vector_type(16)));
typedef _Float16 v8h  __attribute__((ext_vector_type(8)));
typedef _Float16 v4h  __attribute__((ext_vector_type(4)));
typedef float    v8f  __attribute__((ext_vector_type(8)));
typedef float    v4f  __attribute__((ext_vector_type(4)));
typedef int      v4i  __attribute__((ext_vector_type(4)));

typedef __attribute__((address_space(1))) v4i* gp4i_t;   // global int4*
typedef __attribute__((address_space(3))) v4i* lp4i_t;   // LDS int4*

#define WMMA_F16(a, b, c) \
  __builtin_amdgcn_wmma_f32_16x16x32_f16(false, (a), false, (b), (short)0, (c), false, false)

static __device__ __forceinline__ v16h cat8(v8h lo, v8h hi) {
  return __builtin_shufflevector(lo, hi, 0,1,2,3,4,5,6,7,8,9,10,11,12,13,14,15);
}

// N async b128 global->LDS copies with compile-time immediate offsets
// (INST_OFFSET is added to BOTH global and LDS addresses; tracked by ASYNCcnt).
template <int I>
struct QCopy {
  static __device__ __forceinline__ void run(const _Float16* g, _Float16* l) {
    QCopy<I - 1>::run(g, l);
    __builtin_amdgcn_global_load_async_to_lds_b128(
        (gp4i_t)(unsigned long long)g,
        (lp4i_t)(unsigned long long)l, (I - 1) * 16, 0);
  }
};
template <>
struct QCopy<0> {
  static __device__ __forceinline__ void run(const _Float16*, _Float16*) {}
};

// ---------------------------------------------------------------------------
// Kernel 1: QKV = x @ W + b   (f32 in, f16 WMMA compute, f16 out to ws)
//   x: [16384, 768]   W: [768, 2304]   bias: [2304]
//   Qh: [8][2048][768] f16, pre-scaled by 1/sqrt(768)
//   Kh: [8][2048][768] f16
//   VT: [8][768][2048] f16 (V transposed -> contiguous B-frags in attention)
// Workgroup tile 128x64, 8 waves of 32x32, K-step 32 (one WMMA K).
// Global->reg loads for tile k+1 are software-pipelined against WMMA on tile k.
// ---------------------------------------------------------------------------
__global__ __launch_bounds__(256)
void qkv_gemm_f16(const float* __restrict__ x, const float* __restrict__ W,
                  const float* __restrict__ bias,
                  _Float16* __restrict__ Qh, _Float16* __restrict__ Kh,
                  _Float16* __restrict__ VT)
{
  constexpr int AS = 40;   // padded LDS stride (halves): conflict-free frag loads
  constexpr int BS = 40;
  __shared__ _Float16 Ah[128 * AS];   // x tile, row-major (M x K)
  __shared__ _Float16 Bt[64 * BS];    // W tile, transposed (N x K)

  const int t    = threadIdx.x;
  const int lane = t & 31;
  const int w    = t >> 5;
  const int l16  = lane & 15;
  const int lh   = lane >> 4;
  const int m0   = blockIdx.y * 128;
  const int n0   = blockIdx.x * 64;
  const int m_off = (w & 3) * 32;
  const int n_off = (w >> 2) * 32;

  v4f ga[4], gb[2];
  // prologue: load tile k0=0 into registers
  #pragma unroll
  for (int i = 0; i < 4; ++i) {
    int idx = t + 256 * i;                       // 0..1023
    int row = idx >> 3, cb = (idx & 7) << 2;
    ga[i] = *(const v4f*)(x + (size_t)(m0 + row) * 768 + cb);
  }
  #pragma unroll
  for (int i = 0; i < 2; ++i) {
    int idx = t + 256 * i;                       // 0..511
    int kr = idx >> 4, nb = (idx & 15) << 2;
    gb[i] = *(const v4f*)(W + (size_t)kr * 2304 + n0 + nb);
  }

  v8f acc[2][2] = {};

  for (int k0 = 0; k0 < 768; k0 += 32) {
    __syncthreads();                    // previous compute finished reading LDS
    // Stage current tile registers -> LDS (f32 -> f16)
    #pragma unroll
    for (int i = 0; i < 4; ++i) {
      int idx = t + 256 * i;
      int row = idx >> 3, cb = (idx & 7) << 2;
      v4h h = { (_Float16)ga[i][0], (_Float16)ga[i][1],
                (_Float16)ga[i][2], (_Float16)ga[i][3] };
      *(v4h*)(Ah + row * AS + cb) = h;
    }
    #pragma unroll
    for (int i = 0; i < 2; ++i) {
      int idx = t + 256 * i;
      int kr = idx >> 4, nb = (idx & 15) << 2;
      Bt[(nb + 0) * BS + kr] = (_Float16)gb[i][0];
      Bt[(nb + 1) * BS + kr] = (_Float16)gb[i][1];
      Bt[(nb + 2) * BS + kr] = (_Float16)gb[i][2];
      Bt[(nb + 3) * BS + kr] = (_Float16)gb[i][3];
    }
    // Issue next tile's global loads now; they complete during the WMMAs below.
    if (k0 + 32 < 768) {
      int kn = k0 + 32;
      #pragma unroll
      for (int i = 0; i < 4; ++i) {
        int idx = t + 256 * i;
        int row = idx >> 3, cb = (idx & 7) << 2;
        ga[i] = *(const v4f*)(x + (size_t)(m0 + row) * 768 + kn + cb);
      }
      #pragma unroll
      for (int i = 0; i < 2; ++i) {
        int idx = t + 256 * i;
        int kr = idx >> 4, nb = (idx & 15) << 2;
        gb[i] = *(const v4f*)(W + (size_t)(kn + kr) * 2304 + n0 + nb);
      }
    }
    __syncthreads();                    // staged tile visible

    v16h a[2], b[2];
    #pragma unroll
    for (int mt = 0; mt < 2; ++mt) {
      // A frag (16x32 f16): lane needs A[m][16*lh .. +16) contiguous
      const _Float16* p = Ah + (m_off + mt * 16 + l16) * AS + (lh << 4);
      a[mt] = cat8(*(const v8h*)p, *(const v8h*)(p + 8));
    }
    #pragma unroll
    for (int nt = 0; nt < 2; ++nt) {
      // B frag (32x16 f16): lane needs B^T[n][16*lh .. +16) contiguous
      const _Float16* p = Bt + (n_off + nt * 16 + l16) * BS + (lh << 4);
      b[nt] = cat8(*(const v8h*)p, *(const v8h*)(p + 8));
    }
    #pragma unroll
    for (int mt = 0; mt < 2; ++mt)
      #pragma unroll
      for (int nt = 0; nt < 2; ++nt)
        acc[mt][nt] = WMMA_F16(a[mt], b[nt], acc[mt][nt]);
  }

  // Epilogue: uniform branch on lane-independent tile base nb.
  const float qscale = 0.03608439182435161f;  // 1/sqrt(768), folded into Qh
  #pragma unroll
  for (int nt = 0; nt < 2; ++nt) {
    const int nb = n0 + n_off + nt * 16;        // uniform across lanes
    const int n  = nb + l16;
    const float bv = bias[n];
    if (nb < 768) {
      #pragma unroll
      for (int mt = 0; mt < 2; ++mt)
        #pragma unroll
        for (int r = 0; r < 8; ++r) {
          int row = m0 + m_off + mt * 16 + r + (lh << 3);
          Qh[(size_t)row * 768 + n] = (_Float16)((acc[mt][nt][r] + bv) * qscale);
        }
    } else if (nb < 1536) {
      #pragma unroll
      for (int mt = 0; mt < 2; ++mt)
        #pragma unroll
        for (int r = 0; r < 8; ++r) {
          int row = m0 + m_off + mt * 16 + r + (lh << 3);
          Kh[(size_t)row * 768 + (n - 768)] = (_Float16)(acc[mt][nt][r] + bv);
        }
    } else {
      const int d = n - 1536;
      #pragma unroll
      for (int mt = 0; mt < 2; ++mt)
        #pragma unroll
        for (int r = 0; r < 8; ++r) {
          int row = m0 + m_off + mt * 16 + r + (lh << 3);
          int bbv = row >> 11, s = row & 2047;
          VT[((size_t)bbv * 768 + d) * 2048 + s] = (_Float16)(acc[mt][nt][r] + bv);
        }
    }
  }
}

// ---------------------------------------------------------------------------
// Kernel 2: flash attention. 1 WG = 64 query rows. 8 waves:
//   wave w: mt = w&3 (16-row M tile), g = w>>2 (S n-tile group / output D half)
//   Q tile staged once into LDS via GLOBAL_LOAD_ASYNC_TO_LDS_B128 (ASYNCcnt).
//   Per 128-key block: S (64x128) via WMMA, block-wise online softmax,
//   O += P @ V with V pre-transposed (contiguous global B-frags, L2 resident).
// ---------------------------------------------------------------------------
__global__ __launch_bounds__(256)
void flash_attn(const _Float16* __restrict__ Qh, const _Float16* __restrict__ Kh,
                const _Float16* __restrict__ VT, float* __restrict__ out)
{
  __shared__ _Float16 Qs[64][776];     // Q tile (pad 776: conflict-free b128)
  __shared__ float    Ss[64][132];     // scores block (pad 132)
  __shared__ _Float16 Ph[64][136];     // exp(S-m) in f16 (pad 136)
  __shared__ float m_s[64], l_s[64], al_s[64];
  __shared__ float pm[64][4], pl[64][4];

  const int t    = threadIdx.x;
  const int lane = t & 31;
  const int w    = t >> 5;
  const int l16  = lane & 15;
  const int lh   = lane >> 4;
  const int bb   = blockIdx.x >> 5;          // batch
  const int q0   = (blockIdx.x & 31) << 6;   // query tile base
  const _Float16* Qb = Qh + ((size_t)bb * 2048 + q0) * 768;
  const _Float16* Kb = Kh + (size_t)bb * 2048 * 768;
  const _Float16* Vb = VT + (size_t)bb * 768 * 2048;

  const int mt = w & 3;
  const int g  = w >> 2;
  const int d0 = g * 384;                    // this wave's output D slice

  // ---- async-copy Q tile (64x768 f16 = 96 KB) global -> LDS, no VGPR bounce.
  // Thread t owns row t>>2, chunk group (t&3)*24; the per-chunk 16B step is a
  // compile-time immediate applied to both the global and LDS addresses.
  {
    const int qrow = t >> 2;
    const int qcb  = (t & 3) * 192;                  // halves
    QCopy<24>::run(Qb + (size_t)qrow * 768 + qcb, &Qs[qrow][qcb]);
  }
  if (t < 64) { m_s[t] = -3.0e38f; l_s[t] = 0.0f; }
  v8f acc[24] = {};
  __builtin_amdgcn_s_wait_asynccnt(0);       // my copies done
  __syncthreads();                           // everyone's copies done

  for (int j0 = 0; j0 < 2048; j0 += 128) {
    // ---- prefetch next block's K rows / V row-slices toward the WGP
    if (j0 + 128 < 2048) {
      if (w < 4) {
        int kr = j0 + 128 + w * 32 + lane;                   // 128 K rows
        const _Float16* p = Kb + (size_t)kr * 768;
        __builtin_prefetch(p, 0, 1);
        __builtin_prefetch(p + 384, 0, 1);
      } else {
        #pragma unroll
        for (int i = 0; i < 6; ++i) {                        // 768 V rows / 128 lanes
          int dc = (w - 4) * 192 + i * 32 + lane;
          const _Float16* p = Vb + (size_t)dc * 2048 + j0 + 128;
          __builtin_prefetch(p, 0, 1);
          __builtin_prefetch(p + 64, 0, 1);
        }
      }
    }

    // ---- S tiles: wave computes S[mt*16.., (g*4+nt)*16..] over K=768
    v8f st[4] = {};
    for (int kk = 0; kk < 768; kk += 32) {
      const _Float16* pa = &Qs[mt * 16 + l16][kk + (lh << 4)];
      v16h a = cat8(*(const v8h*)pa, *(const v8h*)(pa + 8));
      #pragma unroll
      for (int nt = 0; nt < 4; ++nt) {
        int kr = j0 + (g * 4 + nt) * 16 + l16;
        const _Float16* pb = Kb + (size_t)kr * 768 + kk + (lh << 4);
        v16h b = cat8(*(const v8h*)pb, *(const v8h*)(pb + 8));
        st[nt] = WMMA_F16(a, b, st[nt]);
      }
    }
    #pragma unroll
    for (int nt = 0; nt < 4; ++nt) {
      int col = (g * 4 + nt) * 16 + l16;
      #pragma unroll
      for (int r = 0; r < 8; ++r)
        Ss[mt * 16 + (lh << 3) + r][col] = st[nt][r];
    }
    __syncthreads();

    // ---- block softmax: parallel max / exp / sum across all 256 threads
    {
      int row = t & 63, ch = t >> 6;
      const float* sr = &Ss[row][ch * 32];
      float bm = -3.0e38f;
      #pragma unroll 8
      for (int c = 0; c < 32; ++c) bm = fmaxf(bm, sr[c]);
      pm[row][ch] = bm;
    }
    __syncthreads();
    if (t < 64) {
      float bm = fmaxf(fmaxf(pm[t][0], pm[t][1]), fmaxf(pm[t][2], pm[t][3]));
      float mo = m_s[t], mn = fmaxf(mo, bm);
      al_s[t] = __expf(mo - mn);
      m_s[t]  = mn;
    }
    __syncthreads();
    {
      int row = t & 63, ch = t >> 6;
      float mn = m_s[row];
      float sum = 0.0f;
      #pragma unroll 8
      for (int c = 0; c < 32; ++c) {
        float p = __expf(Ss[row][ch * 32 + c] - mn);
        Ph[row][ch * 32 + c] = (_Float16)p;
        sum += p;
      }
      pl[row][ch] = sum;
    }
    __syncthreads();
    if (t < 64)
      l_s[t] = l_s[t] * al_s[t] + pl[t][0] + pl[t][1] + pl[t][2] + pl[t][3];

    // ---- rescale running accumulators by alpha(row)
    {
      int rb = mt * 16 + (lh << 3);
      #pragma unroll
      for (int r = 0; r < 8; ++r) {
        float a0 = al_s[rb + r];
        #pragma unroll
        for (int n = 0; n < 24; ++n) acc[n][r] *= a0;
      }
    }
    // ---- O += P @ V (V pre-transposed: contiguous b128 frags from L2)
    for (int kk = 0; kk < 128; kk += 32) {
      const _Float16* pa = &Ph[mt * 16 + l16][kk + (lh << 4)];
      v16h a = cat8(*(const v8h*)pa, *(const v8h*)(pa + 8));
      #pragma unroll
      for (int n = 0; n < 24; ++n) {
        int dc = d0 + n * 16 + l16;
        const _Float16* pb = Vb + (size_t)dc * 2048 + j0 + kk + (lh << 4);
        v16h b = cat8(*(const v8h*)pb, *(const v8h*)(pb + 8));
        acc[n] = WMMA_F16(a, b, acc[n]);
      }
    }
    __syncthreads();
  }

  // ---- epilogue: O / l, f32 coalesced stores
  {
    int rb = mt * 16 + (lh << 3);
    #pragma unroll
    for (int r = 0; r < 8; ++r) {
      float inv = 1.0f / l_s[rb + r];
      float* orow = out + (size_t)(bb * 2048 + q0 + rb + r) * 768 + d0;
      #pragma unroll
      for (int n = 0; n < 24; ++n)
        orow[n * 16 + l16] = acc[n][r] * inv;
    }
  }
}

// ---------------------------------------------------------------------------
extern "C" void kernel_launch(void* const* d_in, const int* in_sizes, int n_in,
                              void* d_out, int out_size, void* d_ws, size_t ws_size,
                              hipStream_t stream) {
  const float* x    = (const float*)d_in[0];   // [8,2048,768]
  const float* Wqkv = (const float*)d_in[1];   // [768,2304]
  const float* bqkv = (const float*)d_in[2];   // [2304]
  float* out = (float*)d_out;                  // [8,2048,768]

  const size_t elems = (size_t)8 * 2048 * 768;
  _Float16* Qh = (_Float16*)d_ws;
  _Float16* Kh = Qh + elems;
  _Float16* VT = Kh + elems;

  qkv_gemm_f16<<<dim3(36, 128), dim3(256), 0, stream>>>(x, Wqkv, bqkv, Qh, Kh, VT);
  flash_attn<<<dim3(256), dim3(256), 0, stream>>>(Qh, Kh, VT, out);
}
